// AdaptiveRevRNN_56444460204202
// MI455X (gfx1250) — matrix operand order repaired and checked by reading, not compiled, not run on hardware
//
#include <hip/hip_runtime.h>

// ---------------------------------------------------------------------------
// AdaptiveRevRNN for MI455X (gfx1250): 16-WG cluster, bf16 WMMA recurrence,
// TDM (tensor_load_to_lds) weight staging overlapped with the epilogue.
// B=128, S=2048, F=256, HF=256, DELAY=8, DEPTH=2, T=2056, K=514 (pad 544).
// ---------------------------------------------------------------------------

typedef __attribute__((ext_vector_type(16))) __bf16       v16bf;
typedef __attribute__((ext_vector_type(8)))  float        v8f;
typedef __attribute__((ext_vector_type(4)))  unsigned int v4u;
typedef __attribute__((ext_vector_type(8)))  int          v8i;
typedef __attribute__((ext_vector_type(4)))  int          v4i;

#define B_    128
#define S_    2048
#define F_    256
#define HF_   256
#define T_    2056
#define KT_N  17          // ceil(544/32) K-tiles of 32
#define NWG   16
#define NTHR  256

// workspace layout (bytes)
#define WS_CNT_OFF   0
#define WS_WPACK_OFF 256
#define WPACK_ELEMS  (4*17*32*32*16)              // [w][kt][nt][lane][i] bf16
#define WS_STAG_OFF  (WS_WPACK_OFF + WPACK_ELEMS*2)
#define STAG_ELEMS   (128*256)                    // one A-staging buffer (bf16)

#define BSLICE_BYTES (17*1024)                    // one N-slice of B in LDS

union Frag {
  v16bf  v;
  uint4  q[2];
  __bf16 e[16];
};

__device__ __forceinline__ float fast_tanh(float x) {
  float cx = fminf(fmaxf(x, -9.f), 9.f);
  float e2 = __expf(2.f * cx);
  return (e2 - 1.f) / (e2 + 1.f);
}

// ---------------------------------------------------------------------------
// TDM: DMA one weight N-slice (17 rows x 1024B, row stride 32KB) into LDS.
// 2-D tensor: groups 2/3 zero. data_size=8B -> dims in 8-byte units.
// ---------------------------------------------------------------------------
__device__ __forceinline__ void tdm_load_slice(unsigned lds_addr,
                                               const void* gsrc) {
  unsigned long long ga = (unsigned long long)(uintptr_t)gsrc;
  v4u g0;
  g0.x = 1u;                                          // count=1 (valid), user
  g0.y = lds_addr;                                    // LDS byte address
  g0.z = (unsigned)(ga & 0xFFFFFFFFull);              // global_addr[31:0]
  g0.w = (unsigned)((ga >> 32) & 0x01FFFFFFull)       // global_addr[56:32]
         | (2u << 30);                                // type = 2 ("image")
  v8i g1;
  g1[0] = (int)(3u << 16);       // workgroup_mask=0, data_size=3 (8B)
  g1[1] = (int)(128u << 16);     // tensor_dim0 = 128 (bits 79:48, lo16)
  g1[2] = (int)(17u << 16);      // tensor_dim0 hi=0 ; tensor_dim1 = 17
  g1[3] = (int)(128u << 16);     // tensor_dim1 hi=0 ; tile_dim0 = 128
  g1[4] = 17;                    // tile_dim1 = 17 ; tile_dim2 = 0
  g1[5] = 4096;                  // tensor_dim0_stride = 4096 (32KB / 8B)
  g1[6] = 0;                     // stride hi ; tensor_dim1_stride lo
  g1[7] = 0;
  v4i z4 = {0, 0, 0, 0};
#if __clang_major__ >= 23
  v8i z8 = {0, 0, 0, 0, 0, 0, 0, 0};
  __builtin_amdgcn_tensor_load_to_lds(g0, g1, z4, z4, z8, 0);
#else
  __builtin_amdgcn_tensor_load_to_lds(g0, g1, z4, z4, 0);
#endif
}

// ---------------------------------------------------------------------------
// init: zero the global barrier counter (ws is poisoned, never re-poisoned)
// ---------------------------------------------------------------------------
__global__ void init_ws(unsigned* cnt) {
  if (threadIdx.x == 0 && blockIdx.x == 0) *cnt = 0u;
}

// ---------------------------------------------------------------------------
// prep: pack lp0/lp1 (fp32, [2][514][512]) into WMMA B-fragment order, bf16.
// Tile (w,kt,nt): lane l<16 -> col nt*16+l, K = kt*32 + {0..7, 16..23}
//                 lane l>=16 -> col nt*16+(l-16), K = kt*32 + {8..15, 24..31}
// Rows >= 514 are zero padding (K padded 514 -> 544).
// ---------------------------------------------------------------------------
__global__ void prep_pack(const float* __restrict__ lp0,
                          const float* __restrict__ lp1,
                          __bf16* __restrict__ wpack) {
  int gid = blockIdx.x * blockDim.x + threadIdx.x;
  if (gid >= WPACK_ELEMS) return;
  int i    = gid & 15;
  int r    = gid >> 4;
  int lane = r & 31;      r >>= 5;
  int nt   = r & 31;      r >>= 5;
  int kt   = r % 17;
  int w    = r / 17;                      // 0,1 = lp0 d0,d1 ; 2,3 = lp1 d0,d1
  int krow = kt * 32 + 8 * (lane >> 4) + ((i < 8) ? i : (8 + i));
  int col  = nt * 16 + (lane & 15);
  float v  = 0.f;
  if (krow < 514) {
    const float* lp = (w < 2) ? lp0 : lp1;
    v = lp[(size_t)(w & 1) * 514 * 512 + (size_t)krow * 512 + col];
  }
  wpack[gid] = (__bf16)v;
}

// ---------------------------------------------------------------------------
// main: 16 persistent WGs (one per WGP; cluster-shaped). WG j owns output
// columns [16j,16j+16) of h0/h1 and z-column pair (16j.., 256+16j..).
// ---------------------------------------------------------------------------
__global__ void __launch_bounds__(NTHR, 1)
rnn_main(const float* __restrict__ x, const float* __restrict__ hs,
         const __bf16* __restrict__ wpack, __bf16* __restrict__ stag,
         unsigned* __restrict__ cnt, float* __restrict__ out) {
  __shared__ uint4 sB[2 * 1088];        // 34,816 B: [ntl][kt][lane] B tiles
  __shared__ float sA[128 * 16];        // activated slice (o or intermediate)
  __shared__ float sPS[16 * 16];        // partial column sums
  __shared__ float sMean[16];
  __shared__ float sH0[128 * 16];       // persistent state h0 slice
  __shared__ float sH1[128 * 16];       // persistent state h1 slice

  const int tid  = threadIdx.x;
  const int lane = tid & 31;            // wave32
  const int mw   = tid >> 5;            // wave id == M-tile (8 waves, M=128)
  const int j    = blockIdx.x;          // 0..15

  const unsigned ldsB0 = (unsigned)(uintptr_t)(void*)&sB[0];
  const char*    wbase = (const char*)wpack;

  unsigned barId = 0;
  auto gbar = [&]() {
    ++barId;
    __threadfence();
    __syncthreads();
    if (tid == 0) {
      __builtin_amdgcn_s_cluster_barrier();            // real sync if clustered
      __atomic_fetch_add(cnt, 1u, __ATOMIC_RELEASE);   // fallback L2 barrier
      while (__atomic_load_n(cnt, __ATOMIC_ACQUIRE) < 16u * barId)
        __builtin_amdgcn_s_sleep(2);
    }
    __syncthreads();
  };

  // issue both N-slice DMAs for weight w (wave-uniform; call from wave 0 only)
  auto tdm_issue_w = [&](int w) {
    tdm_load_slice(ldsB0,
                   wbase + (size_t)((w * 17) * 32 + j) * 1024);
    tdm_load_slice(ldsB0 + BSLICE_BYTES,
                   wbase + (size_t)((w * 17) * 32 + 16 + j) * 1024);
  };

  // ---- prologue: h state from hidden_state broadcast; staging buf0.
  // h0 rows are identical across batch -> centered(h0) == 0 exactly.
  for (int e = tid; e < 2048; e += NTHR) {
    int row = e >> 4, col = e & 15;
    sH0[e] = hs[j * 16 + col];
    sH1[e] = hs[256 + j * 16 + col];
    stag[(size_t)row * 256 + j * 16 + col] = (__bf16)0.f;
  }
  if (mw == 0) tdm_issue_w(0);          // weights for sub-step k=0 in flight
  gbar();

  Frag xf[8];   // K-tiles 8..15 : x features (per step)
  Frag pf;      // K-tile 16     : pos features + zero pad

  for (int t = 0; t < T_; ++t) {
    // ---- per-step A fragments for the seq_in half of K
    const int arow = 16 * mw + (lane & 15);
    const float* xrow = x + ((size_t)arow * S_ + (size_t)t) * F_;
    const bool valid = (t < S_);
    const int kbase = (lane & 16) ? 8 : 0;
    for (int kk = 0; kk < 8; ++kk) {
      if (valid) {
        const int f0 = 32 * kk + kbase;
        float4 a = *(const float4*)(xrow + f0);
        float4 b = *(const float4*)(xrow + f0 + 4);
        float4 c = *(const float4*)(xrow + f0 + 16);
        float4 d = *(const float4*)(xrow + f0 + 20);
        xf[kk].e[0]  = (__bf16)a.x; xf[kk].e[1]  = (__bf16)a.y;
        xf[kk].e[2]  = (__bf16)a.z; xf[kk].e[3]  = (__bf16)a.w;
        xf[kk].e[4]  = (__bf16)b.x; xf[kk].e[5]  = (__bf16)b.y;
        xf[kk].e[6]  = (__bf16)b.z; xf[kk].e[7]  = (__bf16)b.w;
        xf[kk].e[8]  = (__bf16)c.x; xf[kk].e[9]  = (__bf16)c.y;
        xf[kk].e[10] = (__bf16)c.z; xf[kk].e[11] = (__bf16)c.w;
        xf[kk].e[12] = (__bf16)d.x; xf[kk].e[13] = (__bf16)d.y;
        xf[kk].e[14] = (__bf16)d.z; xf[kk].e[15] = (__bf16)d.w;
      } else {
        xf[kk].q[0] = make_uint4(0u, 0u, 0u, 0u);
        xf[kk].q[1] = make_uint4(0u, 0u, 0u, 0u);
      }
    }
    pf.q[0] = make_uint4(0u, 0u, 0u, 0u);
    pf.q[1] = make_uint4(0u, 0u, 0u, 0u);
    if (!(lane & 16)) {                          // K=512,513 live in lanes 0-15
      const float idx = (float)(t + 1);
      const float factor = (float)(T_ + 1) * 0.5f;
      pf.e[0] = (__bf16)idx;
      pf.e[1] = (__bf16)((idx - factor) / factor);
    }
    if (t + 1 < S_) __builtin_prefetch(xrow + F_, 0, 0);   // global_prefetch_b8

    for (int s = 0; s < 4; ++s) {                // 4 sequential single_calcs
      const int k = 4 * t + s;
      const __bf16* rbuf = stag + (size_t)(k & 1) * STAG_ELEMS;
      __bf16*       wbuf = stag + (size_t)((k + 1) & 1) * STAG_ELEMS;

      // B tiles for this sub-step were DMA'd by the TDM during the previous
      // epilogue; drain TENSORcnt (no-op for waves that issued nothing).
      __builtin_amdgcn_s_wait_tensorcnt(0);
      __syncthreads();

      // ---- GEMM: (128 x 544) @ (544 x 32) -> this wave: M-tile mw, 2 N-tiles
      v8f accL = {0.f, 0.f, 0.f, 0.f, 0.f, 0.f, 0.f, 0.f};
      v8f accR = {0.f, 0.f, 0.f, 0.f, 0.f, 0.f, 0.f, 0.f};
      for (int kt = 0; kt < KT_N; ++kt) {
        Frag a;
        if (kt < 8) {                            // centered h from staging
          const int kc = 32 * kt + kbase;
          const __bf16* ap = rbuf + (size_t)arow * 256 + kc;
          a.q[0] = *(const uint4*)(ap);
          a.q[1] = *(const uint4*)(ap + 16);
        } else if (kt < 16) {
          a = xf[kt - 8];
        } else {
          a = pf;
        }
        Frag bL, bR;
        const uint4* bp = &sB[kt * 64 + lane * 2];            // ntl = 0 slice
        bL.q[0] = bp[0]; bL.q[1] = bp[1];
        const uint4* bq = &sB[1088 + kt * 64 + lane * 2];     // ntl = 1 slice
        bR.q[0] = bq[0]; bR.q[1] = bq[1];
        accL = __builtin_amdgcn_wmma_f32_16x16x32_bf16(
            false, a.v, false, bL.v, (short)0, accL, false, false);
        accR = __builtin_amdgcn_wmma_f32_16x16x32_bf16(
            false, a.v, false, bR.v, (short)0, accR, false, false);
      }

      // ---- activation: relu6(zL) * tanh(zR); zL/zR share lane+VGPR slot
      for (int r = 0; r < 8; ++r) {
        float zl = accL[r], zr = accR[r];
        float av = fminf(fmaxf(zl, 0.f), 6.f) * fast_tanh(zr);
        int row = 16 * mw + ((lane & 16) ? (8 + r) : r);
        int col = lane & 15;
        sA[row * 16 + col] = av;
      }
      __syncthreads();      // all waves done reading sB and writing sA

      // overlap: DMA next sub-step's weights while we do the epilogue + barrier
      if (mw == 0 && (k + 1) < 4 * T_) tdm_issue_w((k + 1) & 3);

      // ---- residual add + state update (s==1: o1=h1+a ; s==3: o0=h0+a)
      if (s == 1 || s == 3) {
        float* H = (s == 1) ? sH1 : sH0;
        for (int e = tid; e < 2048; e += NTHR) {
          float v = sA[e] + H[e];
          sA[e] = v;
          H[e]  = v;
        }
        __syncthreads();
      }

      // ---- column means over batch (local: we own all 128 rows of our cols)
      {
        int col = tid & 15, chunk = tid >> 4;    // 16 chunks x 8 rows
        float p = 0.f;
        for (int r = 0; r < 8; ++r) p += sA[(chunk * 8 + r) * 16 + col];
        sPS[chunk * 16 + col] = p;
      }
      __syncthreads();
      if (tid < 16) {
        float m = 0.f;
        for (int c2 = 0; c2 < 16; ++c2) m += sPS[c2 * 16 + tid];
        sMean[tid] = m * (1.f / 128.f);
      }
      __syncthreads();

      // ---- centered bf16 slice -> next sub-step's A staging
      for (int e = tid; e < 2048; e += NTHR) {
        int row = e >> 4, col = e & 15;
        wbuf[(size_t)row * 256 + j * 16 + col] = (__bf16)(sA[e] - sMean[col]);
      }
      gbar();
    }
  }

  // ---- epilogue: h_final = [h0 | h1], fp32
  for (int e = tid; e < 2048; e += NTHR) {
    int row = e >> 4, col = e & 15;
    out[(size_t)row * 512 + j * 16 + col]       = sH0[e];
    out[(size_t)row * 512 + 256 + j * 16 + col] = sH1[e];
  }
}

// ---------------------------------------------------------------------------
extern "C" void kernel_launch(void* const* d_in, const int* in_sizes, int n_in,
                              void* d_out, int out_size, void* d_ws,
                              size_t ws_size, hipStream_t stream) {
  (void)in_sizes; (void)n_in; (void)out_size; (void)ws_size;
  const float* x   = (const float*)d_in[0];
  const float* hs  = (const float*)d_in[1];
  const float* lp0 = (const float*)d_in[2];
  const float* lp1 = (const float*)d_in[3];

  char* ws = (char*)d_ws;
  unsigned* cnt  = (unsigned*)(ws + WS_CNT_OFF);
  __bf16*  wpack = (__bf16*)(ws + WS_WPACK_OFF);
  __bf16*  stagb = (__bf16*)(ws + WS_STAG_OFF);

  init_ws<<<1, 32, 0, stream>>>(cnt);
  prep_pack<<<(WPACK_ELEMS + 255) / 256, 256, 0, stream>>>(lp0, lp1, wpack);
  rnn_main<<<NWG, NTHR, 0, stream>>>(x, hs, wpack, stagb, cnt, (float*)d_out);
}
